// MambaCell_31353261261209
// MI455X (gfx1250) — matrix operand order repaired and checked by reading, not compiled, not run on hardware
//
#include <hip/hip_runtime.h>
#include <hip/hip_bf16.h>

// ---------------- problem constants ----------------
#define BATCH   2048
#define D_IN    512
#define D_MODEL 1024
#define ED      2048
#define NSTATE  16
#define KCONV   4
#define DTDIM   64
#define FLAT_H  (ED * NSTATE)               // 32768
#define STATE_W (FLAT_H + ED * (KCONV - 1)) // 38912
#define XZW     (2 * ED)                    // 4096
#define DBCW    128                         // padded dbc row width (96 -> 128)
#define RMS_EPS 1e-5f
#define BK      64                          // GEMM K-step
#define PADK    72                          // LDS row stride in bf16 (64 + 8 pad)

// 1: stage global->LDS with GLOBAL_LOAD_ASYNC_TO_LDS_B128 (ASYNCcnt path)
// 0: fallback register-roundtrip staging
#define USE_ASYNC_LDS 1

typedef __attribute__((ext_vector_type(8)))  float  v8f;
typedef __attribute__((ext_vector_type(16))) __bf16 v16bf;
typedef __attribute__((ext_vector_type(8)))  __bf16 bf16x8;

__device__ __forceinline__ unsigned short f2bf(float f) {
  unsigned int u = __float_as_uint(f);
  u += 0x7FFFu + ((u >> 16) & 1u);          // round-to-nearest-even
  return (unsigned short)(u >> 16);
}
__device__ __forceinline__ float silu_f(float x) { return x / (1.0f + __expf(-x)); }
__device__ __forceinline__ float softplus_f(float x) {
  return (x > 20.0f) ? x : log1pf(__expf(x));
}
__device__ __forceinline__ v16bf frag16(const unsigned short* p, int elemGap) {
  bf16x8 lo = *reinterpret_cast<const bf16x8*>(p);
  bf16x8 hi = *reinterpret_cast<const bf16x8*>(p + elemGap);
  return __builtin_shufflevector(lo, hi, 0, 1, 2, 3, 4, 5, 6, 7,
                                 8, 9, 10, 11, 12, 13, 14, 15);
}

// 16B global -> LDS copy. Async variant bypasses VGPRs (ISA 08 §4: LDS address is taken
// from the VDST VGPR; generic LDS ptr low 32 bits are the LDS offset).
__device__ __forceinline__ void g2l_b128(const unsigned short* g, unsigned short* l) {
#if USE_ASYNC_LDS
  unsigned loff = (unsigned)(uintptr_t)l;
  asm volatile("global_load_async_to_lds_b128 %0, %1, off"
               :: "v"(loff), "v"(g) : "memory");
#else
  *reinterpret_cast<uint4*>(l) = *reinterpret_cast<const uint4*>(g);
#endif
}
__device__ __forceinline__ void g2l_wait() {
#if USE_ASYNC_LDS
  asm volatile("s_wait_asynccnt 0" ::: "memory");
#endif
}

// ---------------- fp32 -> bf16 convert (with optional zero tail) ----------
__global__ void cvt_bf16_kernel(const float* __restrict__ in,
                                unsigned short* __restrict__ out,
                                long nvalid, long ntotal) {
  long i = (long)blockIdx.x * blockDim.x + threadIdx.x;
  if (i < ntotal) out[i] = (i < nvalid) ? f2bf(in[i]) : (unsigned short)0;
}

// ---------------- WMMA GEMM: C[M,N] = A[M,K] * B[N,K]^T ----------------
// A,B bf16 row-major; C fp32. 256 threads = 8 waves. Block tile 128x64,
// wave tile 32x32 (four 16x16 v_wmma_f32_16x16x32_bf16 accumulators).
// Double-buffered LDS, K-step 64 => 8 WMMAs per wave per barrier; staging
// via async global->LDS overlapped with compute. M%128==0, N%64==0, K%64==0.
// MODE 0: plain   MODE 1: softplus(C + colbias)   MODE 2: C + addmat
template <int MODE>
__global__ __launch_bounds__(256) void wmma_gemm_bf16(
    const unsigned short* __restrict__ A, int lda,
    const unsigned short* __restrict__ B, int ldb,
    float* __restrict__ C, int ldc, int Kdim,
    const float* __restrict__ extra) {
  __shared__ unsigned short sA[2][128 * PADK];   // 36.0 KB
  __shared__ unsigned short sB[2][64 * PADK];    // 18.0 KB

  const int tid = threadIdx.x;
  const int m0 = blockIdx.y * 128;
  const int n0 = blockIdx.x * 64;

  // -------- staging geometry (16B chunks, all aligned) --------
  const int ra = tid >> 1, ca = (tid & 1) * 32;  // A: 128r x 64c, 32 bf16/thr
  const int rb = tid >> 2, cb = (tid & 3) * 16;  // B:  64r x 64c, 16 bf16/thr
  const unsigned short* gA = A + (size_t)(m0 + ra) * lda + ca;
  const unsigned short* gB = B + (size_t)(n0 + rb) * ldb + cb;

  auto stage = [&](int buf, int k0) {
#pragma unroll
    for (int c = 0; c < 4; ++c)
      g2l_b128(gA + k0 + c * 8, &sA[buf][ra * PADK + ca + c * 8]);
#pragma unroll
    for (int c = 0; c < 2; ++c)
      g2l_b128(gB + k0 + c * 8, &sB[buf][rb * PADK + cb + c * 8]);
  };

  // -------- wave tile geometry --------
  const int wave = tid >> 5, lane = tid & 31;
  const int half = lane >> 4, l16 = lane & 15;
  const int wm = (wave >> 1) * 32;                // 0,32,64,96
  const int wn = (wave & 1) * 32;                 // 0,32

  v8f acc[2][2] = {};

  auto compute = [&](int buf) {
#pragma unroll
    for (int ks = 0; ks < BK; ks += 32) {
      // B fragments: lanes 0-15 hold K=0..15, 16-31 hold K=16..31 of column N
      v16bf bf0 = frag16(&sB[buf][(wn +  0 + l16) * PADK + ks + half * 16], 8);
      v16bf bf1 = frag16(&sB[buf][(wn + 16 + l16) * PADK + ks + half * 16], 8);
#pragma unroll
      for (int mi = 0; mi < 2; ++mi) {
        // A fragment: half h holds K=h*16+0..15 (matches B)
        v16bf af = frag16(&sA[buf][(wm + mi * 16 + l16) * PADK + ks + half * 16], 8);
        acc[mi][0] = __builtin_amdgcn_wmma_f32_16x16x32_bf16(
            false, af, false, bf0, (short)0, acc[mi][0], false, false);
        acc[mi][1] = __builtin_amdgcn_wmma_f32_16x16x32_bf16(
            false, af, false, bf1, (short)0, acc[mi][1], false, false);
      }
    }
  };

  // -------- pipelined main loop: 1 barrier / 64-K step --------
  const int T = Kdim >> 6;
  stage(0, 0);
  g2l_wait();
  __syncthreads();
  int cur = 0;
  for (int i = 0; i < T; ++i) {
    if (i + 1 < T) stage(cur ^ 1, (i + 1) << 6);  // async, overlaps compute
    compute(cur);
    if (i + 1 < T) {
      g2l_wait();
      __syncthreads();
      cur ^= 1;
    }
  }

  // -------- epilogue --------
#pragma unroll
  for (int ni = 0; ni < 2; ++ni) {
    const int col = n0 + wn + ni * 16 + l16;
#pragma unroll
    for (int mi = 0; mi < 2; ++mi) {
#pragma unroll
      for (int r = 0; r < 8; ++r) {
        const int row = m0 + wm + mi * 16 + half * 8 + r;
        float v = acc[mi][ni][r];
        if (MODE == 1) v = softplus_f(v + extra[col]);
        if (MODE == 2) v += extra[(size_t)row * ldc + col];
        C[(size_t)row * ldc + col] = v;
      }
    }
  }
}

// ---------------- RMSNorm: per-row over D_MODEL, emits bf16 ----------------
__global__ __launch_bounds__(256) void rmsnorm_kernel(
    const float* __restrict__ x, const float* __restrict__ w,
    unsigned short* __restrict__ xn) {
  __shared__ float red[256];
  const int b = blockIdx.x;
  const float* row = x + (size_t)b * D_MODEL;
  float s = 0.0f;
  for (int c = threadIdx.x; c < D_MODEL; c += 256) { float v = row[c]; s += v * v; }
  red[threadIdx.x] = s;
  __syncthreads();
  for (int off = 128; off > 0; off >>= 1) {
    if (threadIdx.x < off) red[threadIdx.x] += red[threadIdx.x + off];
    __syncthreads();
  }
  const float scale = rsqrtf(red[0] * (1.0f / D_MODEL) + RMS_EPS);
  for (int c = threadIdx.x; c < D_MODEL; c += 256)
    xn[(size_t)b * D_MODEL + c] = f2bf(row[c] * scale * w[c]);
}

// ---------------- conv window + SiLU; also writes inputs_new ----------------
__global__ __launch_bounds__(256) void conv_silu_kernel(
    const float* __restrict__ flat_state, const float* __restrict__ xz,
    const float* __restrict__ conv_w, const float* __restrict__ conv_b,
    float* __restrict__ xs_f32, unsigned short* __restrict__ xs_bf,
    float* __restrict__ out_flat /* d_out + B*D_MODEL */) {
  const int idx = blockIdx.x * blockDim.x + threadIdx.x;  // over B*ED
  const int b = idx >> 11, e = idx & (ED - 1);
  const float* inp = flat_state + (size_t)b * STATE_W + FLAT_H + e * 3;
  const float i0 = inp[0], i1 = inp[1], i2 = inp[2];
  const float x1 = xz[(size_t)b * XZW + e];
  const float4 w = *reinterpret_cast<const float4*>(conv_w + e * 4);
  const float xc = i0 * w.x + i1 * w.y + i2 * w.z + x1 * w.w + conv_b[e];
  const float s = silu_f(xc);
  xs_f32[idx] = s;
  xs_bf[idx]  = f2bf(s);
  float* on = out_flat + (size_t)b * STATE_W + FLAT_H + e * 3;
  on[0] = i1; on[1] = i2; on[2] = x1;   // inputs_new
}

// ---------------- SSM state update (HBM-bound, b128 streams) ----------------
__global__ __launch_bounds__(256) void state_update_kernel(
    const float* __restrict__ flat_state, const float* __restrict__ delta,
    const float* __restrict__ dbc, const float* __restrict__ xs,
    const float* __restrict__ xz, const float* __restrict__ A_log,
    const float* __restrict__ D_param,
    float* __restrict__ out_flat, unsigned short* __restrict__ g_bf) {
  const int idx = blockIdx.x * blockDim.x + threadIdx.x;  // over B*ED
  const int b = idx >> 11, e = idx & (ED - 1);
  const float d = delta[idx];
  const float s = xs[idx];
  const float z = xz[(size_t)b * XZW + ED + e];
  const float4* h4  = reinterpret_cast<const float4*>(flat_state + (size_t)b * STATE_W + e * NSTATE);
  float4*       hn4 = reinterpret_cast<float4*>(out_flat + (size_t)b * STATE_W + e * NSTATE);
  const float4* Bm4 = reinterpret_cast<const float4*>(dbc + (size_t)b * DBCW + DTDIM);
  const float4* Cm4 = reinterpret_cast<const float4*>(dbc + (size_t)b * DBCW + DTDIM + NSTATE);
  const float4* Al4 = reinterpret_cast<const float4*>(A_log + e * NSTATE);
  float y = 0.0f;
#pragma unroll
  for (int q = 0; q < 4; ++q) {
    const float4 h = h4[q], Bm = Bm4[q], Cm = Cm4[q], Al = Al4[q];
    float4 hn;
    hn.x = __expf(-__expf(Al.x) * d) * h.x + d * Bm.x * s;
    hn.y = __expf(-__expf(Al.y) * d) * h.y + d * Bm.y * s;
    hn.z = __expf(-__expf(Al.z) * d) * h.z + d * Bm.z * s;
    hn.w = __expf(-__expf(Al.w) * d) * h.w + d * Bm.w * s;
    hn4[q] = hn;
    y += hn.x * Cm.x + hn.y * Cm.y + hn.z * Cm.z + hn.w * Cm.w;
  }
  y += D_param[e] * s;
  g_bf[idx] = f2bf(y * silu_f(z));     // gated input to final GEMM
}

// ---------------- host orchestration ----------------
extern "C" void kernel_launch(void* const* d_in, const int* in_sizes, int n_in,
                              void* d_out, int out_size, void* d_ws, size_t ws_size,
                              hipStream_t stream) {
  const float* x_t       = (const float*)d_in[0];
  const float* flat_st   = (const float*)d_in[1];
  const float* W_in_cell = (const float*)d_in[2];
  const float* norm_w    = (const float*)d_in[3];
  const float* W_in      = (const float*)d_in[4];
  const float* conv_w    = (const float*)d_in[5];
  const float* conv_b    = (const float*)d_in[6];
  const float* W_xproj   = (const float*)d_in[7];
  const float* W_dt      = (const float*)d_in[8];
  const float* b_dt      = (const float*)d_in[9];
  const float* A_log     = (const float*)d_in[10];
  const float* D_param   = (const float*)d_in[11];
  const float* W_out     = (const float*)d_in[12];

  float* out     = (float*)d_out;                 // y_t : [B, D_MODEL]
  float* outFlat = out + (size_t)BATCH * D_MODEL; // flat_new : [B, STATE_W]

  // carve workspace (256B aligned)
  size_t off = 0;
  auto carve = [&](size_t bytes) -> void* {
    off = (off + 255) & ~(size_t)255;
    void* p = (char*)d_ws + off;
    off += bytes;
    return p;
  };
  unsigned short* xt_bf  = (unsigned short*)carve((size_t)BATCH * D_IN * 2);
  unsigned short* Wc_bf  = (unsigned short*)carve((size_t)D_MODEL * D_IN * 2);
  unsigned short* Win_bf = (unsigned short*)carve((size_t)XZW * D_MODEL * 2);
  unsigned short* Wxp_bf = (unsigned short*)carve((size_t)DBCW * ED * 2);   // padded 96->128 rows
  unsigned short* Wdt_bf = (unsigned short*)carve((size_t)ED * DTDIM * 2);
  unsigned short* Wo_bf  = (unsigned short*)carve((size_t)D_MODEL * ED * 2);
  float*          x_res  = (float*)carve((size_t)BATCH * D_MODEL * 4);
  unsigned short* xn_bf  = (unsigned short*)carve((size_t)BATCH * D_MODEL * 2);
  float*          xz     = (float*)carve((size_t)BATCH * XZW * 4);
  float*          xs_f   = (float*)carve((size_t)BATCH * ED * 4);
  unsigned short* xs_bf  = (unsigned short*)carve((size_t)BATCH * ED * 2);
  float*          dbc    = (float*)carve((size_t)BATCH * DBCW * 4);
  unsigned short* dbc_bf = (unsigned short*)carve((size_t)BATCH * DBCW * 2);
  float*          delta  = (float*)carve((size_t)BATCH * ED * 4);
  unsigned short* g_bf   = (unsigned short*)carve((size_t)BATCH * ED * 2);
  (void)ws_size; (void)n_in; (void)in_sizes; (void)out_size;

  auto cvt = [&](const float* src, unsigned short* dst, long nvalid, long ntotal) {
    cvt_bf16_kernel<<<(unsigned)((ntotal + 255) / 256), 256, 0, stream>>>(
        src, dst, nvalid, ntotal);
  };
  cvt(x_t,       xt_bf,  (long)BATCH * D_IN,     (long)BATCH * D_IN);
  cvt(W_in_cell, Wc_bf,  (long)D_MODEL * D_IN,   (long)D_MODEL * D_IN);
  cvt(W_in,      Win_bf, (long)XZW * D_MODEL,    (long)XZW * D_MODEL);
  cvt(W_xproj,   Wxp_bf, (long)96 * ED,          (long)DBCW * ED);  // zero-pad rows 96..127
  cvt(W_dt,      Wdt_bf, (long)ED * DTDIM,       (long)ED * DTDIM);
  cvt(W_out,     Wo_bf,  (long)D_MODEL * ED,     (long)D_MODEL * ED);

  const unsigned gy = BATCH / 128;   // 16

  // 1) x = x_t @ W_in_cell^T  [2048x1024, K=512]
  wmma_gemm_bf16<0><<<dim3(D_MODEL / 64, gy), 256, 0, stream>>>(
      xt_bf, D_IN, Wc_bf, D_IN, x_res, D_MODEL, D_IN, nullptr);

  // 2) xn = rmsnorm(x) -> bf16
  rmsnorm_kernel<<<BATCH, 256, 0, stream>>>(x_res, norm_w, xn_bf);

  // 3) xz = xn @ W_in^T  [2048x4096, K=1024]
  wmma_gemm_bf16<0><<<dim3(XZW / 64, gy), 256, 0, stream>>>(
      xn_bf, D_MODEL, Win_bf, D_MODEL, xz, XZW, D_MODEL, nullptr);

  // 4) conv window + silu; inputs_new -> d_out
  conv_silu_kernel<<<(BATCH * ED) / 256, 256, 0, stream>>>(
      flat_st, xz, conv_w, conv_b, xs_f, xs_bf, outFlat);

  // 5) dbc = xs @ W_xproj_padded^T  [2048x128, K=2048] (cols 96..127 junk-zero)
  wmma_gemm_bf16<0><<<dim3(DBCW / 64, gy), 256, 0, stream>>>(
      xs_bf, ED, Wxp_bf, ED, dbc, DBCW, ED, nullptr);
  cvt(dbc, dbc_bf, (long)BATCH * DBCW, (long)BATCH * DBCW);

  // 6) delta = softplus(dbc[:, :64] @ W_dt^T + b_dt)  [2048x2048, K=64]
  wmma_gemm_bf16<1><<<dim3(ED / 64, gy), 256, 0, stream>>>(
      dbc_bf, DBCW, Wdt_bf, DTDIM, delta, ED, DTDIM, b_dt);

  // 7) SSM state update: h_new -> d_out, g = (y + D*xs)*silu(z) -> bf16
  state_update_kernel<<<(BATCH * ED) / 256, 256, 0, stream>>>(
      flat_st, delta, dbc, xs_f, xz, A_log, D_param, outFlat, g_bf);

  // 8) y_t = g @ W_out^T + residual  [2048x1024, K=2048]
  wmma_gemm_bf16<2><<<dim3(D_MODEL / 64, gy), 256, 0, stream>>>(
      g_bf, ED, Wo_bf, ED, out, D_MODEL, ED, x_res);
}
